// HierarchicalDOMModel_15985868276235
// MI455X (gfx1250) — compile-verified, matrix operand order checked by reading, not run on hardware
//
#include <hip/hip_runtime.h>
#include <hip/hip_bf16.h>
#include <math.h>

typedef __attribute__((ext_vector_type(16))) _Float16 v16h;
typedef __attribute__((ext_vector_type(8)))  _Float16 v8h;
typedef __attribute__((ext_vector_type(8)))  float    v8f;

// ---------------------------------------------------------------------------
// WMMA helpers (CDNA5 gfx1250, wave32, V_WMMA_F32_16X16X32_F16)
// A frag: lane m = lane&15, half = lane>>4; elements 0..7 = k[kb+8h .. +7],
//         elements 8..15 = k[kb+16+8h .. +7]  (two contiguous 16B chunks)
// B frag: pre-swizzled to [frag][lane][16] -> one contiguous 32B load
// C/D   : lane n = lane&15, vgpr r -> m = r + 8*half
// ---------------------------------------------------------------------------
__device__ inline v8f zero8() {
  v8f z = {0.f, 0.f, 0.f, 0.f, 0.f, 0.f, 0.f, 0.f};
  return z;
}
__device__ inline v8h zero8h() {
  v8h z = {(_Float16)0.f, (_Float16)0.f, (_Float16)0.f, (_Float16)0.f,
           (_Float16)0.f, (_Float16)0.f, (_Float16)0.f, (_Float16)0.f};
  return z;
}
__device__ inline v16h cat8(v8h lo, v8h hi) {
  return __builtin_shufflevector(lo, hi, 0, 1, 2, 3, 4, 5, 6, 7, 8, 9, 10, 11,
                                 12, 13, 14, 15);
}

__device__ inline v8f wmma16(v16h a, v16h b, v8f c) {
  return __builtin_amdgcn_wmma_f32_16x16x32_f16(false, a, false, b, (short)0, c,
                                                false, false);
}

__device__ inline int kmap_e(int e, int hlf) {
  int p = e >> 1, j = e & 1;
  int base = (p < 4) ? (2 * p) : (16 + 2 * (p - 4));
  return base + 8 * hlf + j;
}

// A operand from row-major [M][ld] memory, full K=32 slice (vectorized)
__device__ inline v16h fragA_v(const _Float16* p, int ld, int kbase) {
  int lane = threadIdx.x & 31;
  int m = lane & 15, hlf = lane >> 4;
  const _Float16* base = p + (size_t)m * ld + kbase + 8 * hlf;
  v8h lo = *(const v8h*)base;
  v8h hi = *(const v8h*)(base + 16);
  return cat8(lo, hi);
}

// A operand (or B^T operand) with only K=0..7 valid (head-dim 8, zero padded)
__device__ inline v16h fragA8_v(const _Float16* p, int ld, int kbase) {
  int lane = threadIdx.x & 31;
  int m = lane & 15, hlf = lane >> 4;
  v8h lo = zero8h();
  if (hlf == 0) lo = *(const v8h*)(p + (size_t)m * ld + kbase);
  return cat8(lo, zero8h());
}

// B operand from pre-swizzled weights: one contiguous 32B load per lane
__device__ inline v16h fragB_sw(const _Float16* p, int kt, int nt, int Ktiles) {
  int lane = threadIdx.x & 31;
  return *(const v16h*)(p + (((size_t)nt * Ktiles + kt) * 32 + lane) * 16);
}

// B operand for A@V: V stored transposed [64 chan][128 seq]; n<8 valid
__device__ inline v16h fragBt8(const _Float16* vt, int h, int kbase) {
  int lane = threadIdx.x & 31;
  int n = lane & 15, hlf = lane >> 4;
  v8h lo = zero8h(), hi = zero8h();
  if (n < 8) {
    const _Float16* base = vt + (size_t)(h * 8 + n) * 128 + kbase + 8 * hlf;
    lo = *(const v8h*)base;
    hi = *(const v8h*)(base + 16);
  }
  return cat8(lo, hi);
}

// ---------------------------------------------------------------------------
// weight conversion: f32 row-major [mats][Ksrc][N] -> f16 fragment-swizzled
// dst layout per matrix: [(nt*Ktiles+kt)][lane][e], K zero-padded to Kpad
// ---------------------------------------------------------------------------
__global__ void k_cvt_swz(const float* __restrict__ src,
                          _Float16* __restrict__ dst, int Ksrc, int Kpad, int N,
                          int mats) {
  int Ktiles = Kpad / 32, Ntiles = N / 16;
  int per = Ktiles * Ntiles * 512;
  int total = mats * per;
  int i = blockIdx.x * 256 + threadIdx.x;
  if (i >= total) return;
  int mat = i / per, r = i % per;
  int fi = r / 512, r2 = r % 512;
  int nt = fi / Ktiles, kt = fi % Ktiles;
  int lane = r2 / 16, e = r2 % 16;
  int n = nt * 16 + (lane & 15);
  int k = kt * 32 + kmap_e(e, lane >> 4);
  float v = (k < Ksrc) ? src[((size_t)mat * Ksrc + k) * N + n] : 0.f;
  dst[i] = (_Float16)v;
}

// initialize padded x tensor to pos_b broadcast (pad rows keep this value)
__global__ void k_xinit(float* __restrict__ x, const float* __restrict__ pos_b,
                        int n) {
  int i = blockIdx.x * 256 + threadIdx.x;
  if (i < n) x[i] = pos_b[i & 63];
}

// ---------------------------------------------------------------------------
// Kernel 1: fused pulse feature build + 3-layer MLP (WMMA) + mean/max DOM pool
// one wave per DOM (32 pulses = two 16-row WMMA tiles); pe never hits HBM
// ---------------------------------------------------------------------------
__global__ void __launch_bounds__(256) k_pulse_dom(
    const float* __restrict__ pf, const int* __restrict__ p2d,
    const int* __restrict__ npd, const float* __restrict__ dpos,
    const _Float16* __restrict__ w1, const _Float16* __restrict__ w2,
    const _Float16* __restrict__ w3, const float* __restrict__ b1,
    const float* __restrict__ b2, const float* __restrict__ b3,
    _Float16* __restrict__ domf) {
  extern __shared__ char smem[];
  const int tid = threadIdx.x, lane = tid & 31, wave = tid >> 5;
  const int hlf = lane >> 4, n0 = lane & 15;
  char* wb = smem + wave * (1024 + 2048 + 8192);
  _Float16* A0 = (_Float16*)wb;           // [16][32] staged features (K padded)
  _Float16* Hh = (_Float16*)(wb + 1024);  // [16][64] hidden (in-place reuse)
  float* PE = (float*)(wb + 3072);        // [64 chan][32 pulse] channel-major
  const int dom = blockIdx.x * 8 + wave;

  for (int t = 0; t < 2; ++t) {
    const int p0 = dom * 32 + t * 16;
    // stage features: zero-pad k>=8, fill 8 real features
    for (int i = lane; i < 16 * 24; i += 32)
      A0[(i / 24) * 32 + 8 + (i % 24)] = (_Float16)0.0f;
    for (int i = lane; i < 16 * 8; i += 32) {
      int row = i >> 3, col = i & 7, p = p0 + row;
      float v;
      if (col < 4) v = pf[(size_t)p * 4 + col];
      else if (col < 7) { int dd = p2d[p]; v = dpos[(size_t)dd * 3 + (col - 4)]; }
      else v = log1pf((float)npd[p]);
      A0[row * 32 + col] = (_Float16)v;
    }
    __syncthreads();
    // layer 1: [16x32(pad)] @ [32x64]
    {
      v16h a = fragA_v(A0, 32, 0);
#pragma unroll
      for (int nt = 0; nt < 4; ++nt) {
        v8f acc = wmma16(a, fragB_sw(w1, 0, nt, 1), zero8());
        int n = nt * 16 + n0;
        float bb = b1[n];
#pragma unroll
        for (int r = 0; r < 8; ++r) {
          float v = acc[r] + bb;
          Hh[(r + 8 * hlf) * 64 + n] = (_Float16)(v > 0.f ? v : 0.f);
        }
      }
    }
    __syncthreads();
    // layer 2: [16x64] @ [64x64] (A frags preloaded -> in-place update of Hh)
    {
      v16h a0 = fragA_v(Hh, 64, 0), a1 = fragA_v(Hh, 64, 32);
#pragma unroll
      for (int nt = 0; nt < 4; ++nt) {
        v8f acc = wmma16(a0, fragB_sw(w2, 0, nt, 2), zero8());
        acc = wmma16(a1, fragB_sw(w2, 1, nt, 2), acc);
        int n = nt * 16 + n0;
        float bb = b2[n];
#pragma unroll
        for (int r = 0; r < 8; ++r) {
          float v = acc[r] + bb;
          Hh[(r + 8 * hlf) * 64 + n] = (_Float16)(v > 0.f ? v : 0.f);
        }
      }
    }
    __syncthreads();
    // layer 3 -> PE (channel-major, vectorized v8f store per lane)
    {
      v16h a0 = fragA_v(Hh, 64, 0), a1 = fragA_v(Hh, 64, 32);
#pragma unroll
      for (int nt = 0; nt < 4; ++nt) {
        v8f acc = wmma16(a0, fragB_sw(w3, 0, nt, 2), zero8());
        acc = wmma16(a1, fragB_sw(w3, 1, nt, 2), acc);
        int n = nt * 16 + n0;
        float bb = b3[n];
        v8f pv;
#pragma unroll
        for (int r = 0; r < 8; ++r) pv[r] = acc[r] + bb;
        *(v8f*)(PE + (size_t)n * 32 + t * 16 + 8 * hlf) = pv;
      }
    }
    __syncthreads();
  }
  // mean+max pool over the 32 pulses of this DOM -> concat [mean|max] (128) f16
  for (int c = lane; c < 64; c += 32) {
    float s = 0.f, mx = -3.4e38f;
    for (int r = 0; r < 32; ++r) {
      float v = PE[c * 32 + r];
      s += v;
      mx = fmaxf(mx, v);
    }
    domf[(size_t)dom * 128 + c] = (_Float16)(s * (1.f / 32.f));
    domf[(size_t)dom * 128 + 64 + c] = (_Float16)mx;
  }
}

// ---------------------------------------------------------------------------
// Kernel 2: DOM projection (concat@pool_W) + positional proj, scatter into x
// one wave per 16 DOMs
// ---------------------------------------------------------------------------
__global__ void __launch_bounds__(32) k_dom_project(
    const _Float16* __restrict__ domf, const _Float16* __restrict__ poolw,
    const float* __restrict__ pool_b, const float* __restrict__ dpos,
    const float* __restrict__ posW, const float* __restrict__ pos_b,
    const int* __restrict__ d2e, float* __restrict__ x, int DPE, int MAXD) {
  const int lane = threadIdx.x & 31, hlf = lane >> 4, n0 = lane & 15;
  const int d0 = blockIdx.x * 16;
  const _Float16* Arow = domf + (size_t)d0 * 128;
  v16h a[4];
#pragma unroll
  for (int kt = 0; kt < 4; ++kt) a[kt] = fragA_v(Arow, 128, kt * 32);
#pragma unroll
  for (int nt = 0; nt < 4; ++nt) {
    v8f acc = zero8();
#pragma unroll
    for (int kt = 0; kt < 4; ++kt)
      acc = wmma16(a[kt], fragB_sw(poolw, kt, nt, 4), acc);
    int c = nt * 16 + n0;
    float bb = pool_b[c] + pos_b[c];
#pragma unroll
    for (int r = 0; r < 8; ++r) {
      int dom = d0 + r + 8 * hlf;
      int ev = d2e[dom];
      int s = dom - ev * DPE;
      float px = dpos[(size_t)dom * 3 + 0];
      float py = dpos[(size_t)dom * 3 + 1];
      float pz = dpos[(size_t)dom * 3 + 2];
      float v = acc[r] + bb + px * posW[0 * 64 + c] + py * posW[1 * 64 + c] +
                pz * posW[2 * 64 + c];
      x[((size_t)ev * MAXD + s) * 64 + c] = v;
    }
  }
}

// ---------------------------------------------------------------------------
// Kernel 3: fused 4-layer transformer, one block = one event, x lives in LDS
// ---------------------------------------------------------------------------
struct TfArgs {
  float* x;        // [B][128][64] fp32
  const int* cnt;  // [B]
  const _Float16 *wq, *wk, *wv, *wo, *f1, *f2;  // f16 swizzled weights
  const float *bq, *bk, *bv, *bo;
  const float *ln1s, *ln1b, *fb1, *fb2, *ln2s, *ln2b;
  _Float16* ev;  // [B][64] pooled event embedding (f16)
  int L;
};

__device__ inline void layer_norm_row(float* X32, _Float16* X16, int row,
                                      const float* sc, const float* bi) {
  float mean = 0.f;
  for (int c = 0; c < 64; ++c) mean += X32[row * 64 + c];
  mean *= (1.0f / 64.0f);
  float var = 0.f;
  for (int c = 0; c < 64; ++c) {
    float d = X32[row * 64 + c] - mean;
    var += d * d;
  }
  var *= (1.0f / 64.0f);
  float rstd = rsqrtf(var + 1e-5f);
  for (int c = 0; c < 64; ++c) {
    float v = (X32[row * 64 + c] - mean) * rstd * sc[c] + bi[c];
    X32[row * 64 + c] = v;
    X16[row * 64 + c] = (_Float16)v;
  }
}

__global__ void __launch_bounds__(256) k_transformer(TfArgs A) {
  extern __shared__ char smem[];
  float* X32 = (float*)smem;                    // [128][64]   32768 B
  _Float16* X16 = (_Float16*)(smem + 32768);    // [128][64]   16384 B
  _Float16* R1 = (_Float16*)(smem + 49152);     // 65536 B: Q|K|Vt then FF h1
  _Float16* Q = R1;                             // [128][64] row-major
  _Float16* Kk = R1 + 8192;                     // [128][64] row-major
  _Float16* Vt = R1 + 16384;                    // [64][128] transposed
  _Float16* O16 = (_Float16*)(smem + 114688);   // [128][64]   16384 B
  _Float16* PROB = (_Float16*)(smem + 131072);  // [8 waves][16][128] 32768 B

  const int tid = threadIdx.x, lane = tid & 31, wave = tid >> 5;
  const int hlf = lane >> 4, n0 = lane & 15;
  const int b = blockIdx.x;
  const int cnt = A.cnt[b];
  float* xg = A.x + (size_t)b * 128 * 64;

  for (int i = tid; i < 128 * 64; i += 256) {
    float v = xg[i];
    X32[i] = v;
    X16[i] = (_Float16)v;
  }
  __syncthreads();

  for (int l = 0; l < A.L; ++l) {
    // ---- QKV projections: wave w owns row-tile w ----
    {
      const _Float16* wm[3] = {A.wq + l * 4096, A.wk + l * 4096,
                               A.wv + l * 4096};
      const float* bi[3] = {A.bq + l * 64, A.bk + l * 64, A.bv + l * 64};
      _Float16* ds[2] = {Q, Kk};
      const int mt = wave;
      v16h a0 = fragA_v(X16 + mt * 16 * 64, 64, 0);
      v16h a1 = fragA_v(X16 + mt * 16 * 64, 64, 32);
      for (int m3 = 0; m3 < 3; ++m3) {
#pragma unroll
        for (int nt = 0; nt < 4; ++nt) {
          v8f acc = wmma16(a0, fragB_sw(wm[m3], 0, nt, 2), zero8());
          acc = wmma16(a1, fragB_sw(wm[m3], 1, nt, 2), acc);
          int n = nt * 16 + n0;
          float bb = bi[m3][n];
          if (m3 == 2) {
            // V stored transposed [chan][seq]: one packed 16B store per lane
            v8h pk;
#pragma unroll
            for (int r = 0; r < 8; ++r) pk[r] = (_Float16)(acc[r] + bb);
            *(v8h*)(Vt + (size_t)n * 128 + mt * 16 + 8 * hlf) = pk;
          } else {
#pragma unroll
            for (int r = 0; r < 8; ++r)
              ds[m3][(mt * 16 + r + 8 * hlf) * 64 + n] =
                  (_Float16)(acc[r] + bb);
          }
        }
      }
    }
    __syncthreads();
    // ---- attention: wave == head (H=8, hd=8, K zero-padded 8->32) ----
    {
      const int h = wave;
      for (int mt = 0; mt < 8; ++mt) {
        v8f s[8];
        v16h aq = fragA8_v(Q + mt * 16 * 64, 64, h * 8);
#pragma unroll
        for (int nt = 0; nt < 8; ++nt) {
          v16h bkf = fragA8_v(Kk + nt * 16 * 64, 64, h * 8);  // B^T (keys)
          s[nt] = wmma16(aq, bkf, zero8());
        }
#pragma unroll
        for (int nt = 0; nt < 8; ++nt) {
          float msk = (nt * 16 + n0 < cnt) ? 0.0f : -1e9f;
#pragma unroll
          for (int r = 0; r < 8; ++r)
            s[nt][r] = s[nt][r] * 0.35355339059f + msk;
        }
#pragma unroll
        for (int r = 0; r < 8; ++r) {
          float mx = -3.4e38f;
#pragma unroll
          for (int nt = 0; nt < 8; ++nt) mx = fmaxf(mx, s[nt][r]);
#pragma unroll
          for (int o = 8; o >= 1; o >>= 1) mx = fmaxf(mx, __shfl_xor(mx, o, 32));
          float sum = 0.f;
#pragma unroll
          for (int nt = 0; nt < 8; ++nt) {
            float e = __expf(s[nt][r] - mx);
            s[nt][r] = e;
            sum += e;
          }
#pragma unroll
          for (int o = 8; o >= 1; o >>= 1) sum += __shfl_xor(sum, o, 32);
          float inv = 1.0f / sum;
          int m = r + 8 * hlf;
#pragma unroll
          for (int nt = 0; nt < 8; ++nt)
            PROB[wave * 2048 + m * 128 + nt * 16 + n0] =
                (_Float16)(s[nt][r] * inv);
        }
        __syncthreads();
        v8f oacc = zero8();
#pragma unroll
        for (int kt = 0; kt < 4; ++kt) {
          v16h ap = fragA_v(PROB + wave * 2048, 128, kt * 32);
          v16h bvf = fragBt8(Vt, h, kt * 32);
          oacc = wmma16(ap, bvf, oacc);
        }
        if (n0 < 8) {
#pragma unroll
          for (int r = 0; r < 8; ++r)
            O16[(mt * 16 + r + 8 * hlf) * 64 + h * 8 + n0] = (_Float16)oacc[r];
        }
        __syncthreads();
      }
    }
    // ---- output projection + residual (in-place into X32) ----
    {
      const int mt = wave;
      const _Float16* wo = A.wo + l * 4096;
      v16h a0 = fragA_v(O16 + mt * 16 * 64, 64, 0);
      v16h a1 = fragA_v(O16 + mt * 16 * 64, 64, 32);
#pragma unroll
      for (int nt = 0; nt < 4; ++nt) {
        v8f acc = wmma16(a0, fragB_sw(wo, 0, nt, 2), zero8());
        acc = wmma16(a1, fragB_sw(wo, 1, nt, 2), acc);
        int n = nt * 16 + n0;
        float bb = A.bo[l * 64 + n];
#pragma unroll
        for (int r = 0; r < 8; ++r)
          X32[(mt * 16 + r + 8 * hlf) * 64 + n] += acc[r] + bb;
      }
    }
    __syncthreads();
    if (tid < 128)
      layer_norm_row(X32, X16, tid, A.ln1s + l * 64, A.ln1b + l * 64);
    __syncthreads();
    // ---- FF1: h1 = relu(X @ W1 + b1) into R1 [128][256] ----
    {
      const int mt = wave;
      const _Float16* w1 = A.f1 + l * 16384;
      v16h a0 = fragA_v(X16 + mt * 16 * 64, 64, 0);
      v16h a1 = fragA_v(X16 + mt * 16 * 64, 64, 32);
#pragma unroll
      for (int nt = 0; nt < 16; ++nt) {
        v8f acc = wmma16(a0, fragB_sw(w1, 0, nt, 2), zero8());
        acc = wmma16(a1, fragB_sw(w1, 1, nt, 2), acc);
        int n = nt * 16 + n0;
        float bb = A.fb1[l * 256 + n];
#pragma unroll
        for (int r = 0; r < 8; ++r) {
          float v = acc[r] + bb;
          R1[(mt * 16 + r + 8 * hlf) * 256 + n] = (_Float16)(v > 0.f ? v : 0.f);
        }
      }
    }
    __syncthreads();
    // ---- FF2 + residual ----
    {
      const int mt = wave;
      const _Float16* w2 = A.f2 + l * 16384;
#pragma unroll
      for (int nt = 0; nt < 4; ++nt) {
        v8f acc = zero8();
#pragma unroll
        for (int kt = 0; kt < 8; ++kt) {
          v16h a = fragA_v(R1 + mt * 16 * 256, 256, kt * 32);
          acc = wmma16(a, fragB_sw(w2, kt, nt, 8), acc);
        }
        int n = nt * 16 + n0;
        float bb = A.fb2[l * 64 + n];
#pragma unroll
        for (int r = 0; r < 8; ++r)
          X32[(mt * 16 + r + 8 * hlf) * 64 + n] += acc[r] + bb;
      }
    }
    __syncthreads();
    if (tid < 128)
      layer_norm_row(X32, X16, tid, A.ln2s + l * 64, A.ln2b + l * 64);
    __syncthreads();
  }
  // ---- masked mean pool over valid rows ----
  if (tid < 64) {
    float s = 0.f;
    for (int r = 0; r < cnt; ++r) s += X32[r * 64 + tid];
    A.ev[(size_t)b * 64 + tid] = (_Float16)(s / fmaxf((float)cnt, 1.0f));
  }
}

// ---------------------------------------------------------------------------
// Kernel 4: head: relu(ev @ W1 + b1) @ W2 + b2, normalize. One wave / 16 rows
// ---------------------------------------------------------------------------
__global__ void __launch_bounds__(32) k_head(const _Float16* __restrict__ ev,
                                             const _Float16* __restrict__ w1,
                                             const float* __restrict__ b1,
                                             const float* __restrict__ w2,
                                             const float* __restrict__ b2,
                                             float* __restrict__ out) {
  __shared__ float H[16 * 128];
  const int lane = threadIdx.x & 31, hlf = lane >> 4, n0 = lane & 15;
  const int m0 = blockIdx.x * 16;
  v16h a0 = fragA_v(ev + (size_t)m0 * 64, 64, 0);
  v16h a1 = fragA_v(ev + (size_t)m0 * 64, 64, 32);
#pragma unroll
  for (int nt = 0; nt < 8; ++nt) {
    v8f acc = wmma16(a0, fragB_sw(w1, 0, nt, 2), zero8());
    acc = wmma16(a1, fragB_sw(w1, 1, nt, 2), acc);
    int n = nt * 16 + n0;
    float bb = b1[n];
#pragma unroll
    for (int r = 0; r < 8; ++r) {
      float v = acc[r] + bb;
      H[(r + 8 * hlf) * 128 + n] = v > 0.f ? v : 0.f;
    }
  }
  __syncthreads();
  if (lane < 16) {
    int row = m0 + lane;
    float d0 = b2[0], d1 = b2[1], d2 = b2[2];
    for (int k = 0; k < 128; ++k) {
      float h = H[lane * 128 + k];
      d0 += h * w2[k * 3 + 0];
      d1 += h * w2[k * 3 + 1];
      d2 += h * w2[k * 3 + 2];
    }
    float inv = rsqrtf(d0 * d0 + d1 * d1 + d2 * d2);
    out[(size_t)row * 3 + 0] = d0 * inv;
    out[(size_t)row * 3 + 1] = d1 * inv;
    out[(size_t)row * 3 + 2] = d2 * inv;
  }
}

// ---------------------------------------------------------------------------
// host launcher
// ---------------------------------------------------------------------------
extern "C" void kernel_launch(void* const* d_in, const int* in_sizes, int n_in,
                              void* d_out, int out_size, void* d_ws,
                              size_t ws_size, hipStream_t stream) {
  const float* pf = (const float*)d_in[0];
  const int* p2d = (const int*)d_in[1];
  const int* npd = (const int*)d_in[3];
  const float* dpos = (const float*)d_in[4];
  const int* d2e = (const int*)d_in[5];
  const int* evc = (const int*)d_in[6];
  const float* pe_W1 = (const float*)d_in[10];
  const float* pe_b1 = (const float*)d_in[11];
  const float* pe_W2 = (const float*)d_in[12];
  const float* pe_b2 = (const float*)d_in[13];
  const float* pe_W3 = (const float*)d_in[14];
  const float* pe_b3 = (const float*)d_in[15];
  const float* pool_W = (const float*)d_in[16];
  const float* pool_b = (const float*)d_in[17];
  const float* pos_W = (const float*)d_in[18];
  const float* pos_b = (const float*)d_in[19];
  const float* Wq = (const float*)d_in[20];
  const float* bq = (const float*)d_in[21];
  const float* Wk = (const float*)d_in[22];
  const float* bk = (const float*)d_in[23];
  const float* Wv = (const float*)d_in[24];
  const float* bv = (const float*)d_in[25];
  const float* Wo = (const float*)d_in[26];
  const float* bo = (const float*)d_in[27];
  const float* ln1s = (const float*)d_in[28];
  const float* ln1b = (const float*)d_in[29];
  const float* fW1 = (const float*)d_in[30];
  const float* fb1 = (const float*)d_in[31];
  const float* fW2 = (const float*)d_in[32];
  const float* fb2 = (const float*)d_in[33];
  const float* ln2s = (const float*)d_in[34];
  const float* ln2b = (const float*)d_in[35];
  const float* hW1 = (const float*)d_in[36];
  const float* hb1 = (const float*)d_in[37];
  const float* hW2 = (const float*)d_in[38];
  const float* hb2 = (const float*)d_in[39];

  const int P = in_sizes[0] / 4;  // 819200
  const int D = in_sizes[4] / 3;  // 25600
  const int B = in_sizes[6];      // 256
  const int E = 64, MAXD = 128, L = 4, FF = 256, HH = 128;
  const int DPE = D / B;  // 100

  // workspace carve
  char* ws = (char*)d_ws;
  size_t off = 0;
#define CARVE(bytes) \
  (ws + off);        \
  off = (off + (size_t)(bytes) + 255) & ~(size_t)255
  _Float16* w1p = (_Float16*)CARVE(32 * 64 * 2);       // 8->32 padded, swz
  _Float16* w2h = (_Float16*)CARVE(E * E * 2);         // swz
  _Float16* w3h = (_Float16*)CARVE(E * E * 2);         // swz
  _Float16* pwh = (_Float16*)CARVE(2 * E * E * 2);     // swz
  _Float16* wqh = (_Float16*)CARVE(L * E * E * 2);     // swz
  _Float16* wkh = (_Float16*)CARVE(L * E * E * 2);     // swz
  _Float16* wvh = (_Float16*)CARVE(L * E * E * 2);     // swz
  _Float16* woh = (_Float16*)CARVE(L * E * E * 2);     // swz
  _Float16* f1h = (_Float16*)CARVE(L * E * FF * 2);    // swz
  _Float16* f2h = (_Float16*)CARVE(L * FF * E * 2);    // swz
  _Float16* hw1h = (_Float16*)CARVE(E * HH * 2);       // swz
  _Float16* domf = (_Float16*)CARVE((size_t)D * 2 * E * 2);
  float* xbuf = (float*)CARVE((size_t)B * MAXD * E * 4);
  _Float16* evh = (_Float16*)CARVE((size_t)B * E * 2);
#undef CARVE

  // weight conversions: f32 -> f16, fragment-swizzled (pe_W1 K padded 8->32)
#define SWZ(src, dst, Ksrc, Kpad, N, mats)                                   \
  k_cvt_swz<<<(((mats) * ((Kpad) / 32) * ((N) / 16) * 512) + 255) / 256, 256, \
              0, stream>>>(src, dst, Ksrc, Kpad, N, mats)
  SWZ(pe_W1, w1p, 8, 32, 64, 1);
  SWZ(pe_W2, w2h, 64, 64, 64, 1);
  SWZ(pe_W3, w3h, 64, 64, 64, 1);
  SWZ(pool_W, pwh, 128, 128, 64, 1);
  SWZ(Wq, wqh, 64, 64, 64, 4);
  SWZ(Wk, wkh, 64, 64, 64, 4);
  SWZ(Wv, wvh, 64, 64, 64, 4);
  SWZ(Wo, woh, 64, 64, 64, 4);
  SWZ(fW1, f1h, 64, 64, 256, 4);
  SWZ(fW2, f2h, 256, 256, 64, 4);
  SWZ(hW1, hw1h, 64, 64, 128, 1);
#undef SWZ

  // fused pulse MLP + DOM pooling: one wave per DOM
  size_t sm1 = (size_t)8 * (1024 + 2048 + 8192);
  k_pulse_dom<<<D / 8, 256, sm1, stream>>>(pf, p2d, npd, dpos, w1p, w2h, w3h,
                                           pe_b1, pe_b2, pe_b3, domf);

  // init padded x with pos_b (pad rows keep it), then scatter real DOM rows
  int nx = B * MAXD * E;
  k_xinit<<<(nx + 255) / 256, 256, 0, stream>>>(xbuf, pos_b, nx);
  k_dom_project<<<D / 16, 32, 0, stream>>>(domf, pwh, pool_b, dpos, pos_W,
                                           pos_b, d2e, xbuf, DPE, MAXD);

  // fused 4-layer transformer, one event per block, 160KB dynamic LDS
  TfArgs targ;
  targ.x = xbuf;
  targ.cnt = evc;
  targ.wq = wqh;
  targ.wk = wkh;
  targ.wv = wvh;
  targ.wo = woh;
  targ.f1 = f1h;
  targ.f2 = f2h;
  targ.bq = bq;
  targ.bk = bk;
  targ.bv = bv;
  targ.bo = bo;
  targ.ln1s = ln1s;
  targ.ln1b = ln1b;
  targ.fb1 = fb1;
  targ.fb2 = fb2;
  targ.ln2s = ln2s;
  targ.ln2b = ln2b;
  targ.ev = evh;
  targ.L = L;
  k_transformer<<<B, 256, 163840, stream>>>(targ);

  // directional head
  k_head<<<B / 16, 32, 0, stream>>>(evh, hw1h, hb1, hW2, hb2, (float*)d_out);

  (void)n_in;
  (void)out_size;
  (void)ws_size;
  (void)P;
}
